// MyGCN_29205777612871
// MI455X (gfx1250) — matrix operand order repaired and checked by reading, not compiled, not run on hardware
//
#include <hip/hip_runtime.h>
#include <math.h>

// ---------------------------------------------------------------------------
// Types for CDNA5 WMMA
// ---------------------------------------------------------------------------
typedef __attribute__((ext_vector_type(16))) __bf16 v16bf;
typedef __attribute__((ext_vector_type(8)))  float  v8f;

union ABfrag { v16bf v; unsigned int u[8]; };

__device__ inline unsigned short f32_to_bf16(float f) {
  unsigned int u = __float_as_uint(f);
  unsigned int r = u + 0x7FFFu + ((u >> 16) & 1u);   // round-to-nearest-even
  return (unsigned short)(r >> 16);
}

__device__ inline void atomicMaxFloat(float* addr, float value) {
  if (value >= 0.f) atomicMax((int*)addr, __float_as_int(value));
  else              atomicMin((unsigned int*)addr, __float_as_uint(value));
}

// ---------------------------------------------------------------------------
// GEMM: C[M,128] = A[M,128] @ B[128,128], bf16 WMMA with f32 accumulate.
//   Bt : pre-converted bf16 B, stored transposed [n][k].
//   MODE 0: plain           C = A @ B
//   MODE 1: fused edge op   C = (A * rowScale[row]) @ B + addTab[addIdx[row],:]
// Each wave owns a 16(M)x128(N) strip: 8 accumulator tiles held in registers,
// K-outer / N-inner so consecutive WMMAs reuse the same A fragment, and the
// epilogue is row-major (1 address + 1 guard per row, 8 stores w/ imm offsets).
// ---------------------------------------------------------------------------
template <int MODE>
__global__ __launch_bounds__(256) void gemm128_bf16(
    const float* __restrict__ A, const unsigned short* __restrict__ Bt,
    float* __restrict__ C, int M,
    const float* __restrict__ rowScale,
    const float* __restrict__ addTab,
    const int* __restrict__ addIdx)
{
  __shared__ unsigned short Alds[128 * 128];
  __shared__ unsigned short Blds[128 * 128];
  const int tid = threadIdx.x;
  const int m0  = blockIdx.x * 128;

  // B tile: already bf16, already [n][k]; copy as b128
  {
    const uint4* Bu = (const uint4*)Bt;
    uint4* Bl = (uint4*)Blds;
#pragma unroll
    for (int i = 0; i < 8; ++i) Bl[tid + i * 256] = Bu[tid + i * 256];
  }
  // A tile: float4 loads, fp32 -> bf16 (optional per-row scale), zero-pad past M
  {
    const float4* Af = (const float4*)A;
#pragma unroll
    for (int it = 0; it < 16; ++it) {
      int i4  = tid + it * 256;          // 0..4095
      int row = i4 >> 5;                 // 32 float4 per row
      int c4  = i4 & 31;
      int gr  = m0 + row;
      float4 v = make_float4(0.f, 0.f, 0.f, 0.f);
      if (gr < M) {
        v = Af[(size_t)gr * 32 + c4];
        if (MODE == 1) {
          float sc = rowScale[gr];
          v.x *= sc; v.y *= sc; v.z *= sc; v.w *= sc;
        }
      }
      uint2 pk;
      pk.x = (unsigned)f32_to_bf16(v.x) | ((unsigned)f32_to_bf16(v.y) << 16);
      pk.y = (unsigned)f32_to_bf16(v.z) | ((unsigned)f32_to_bf16(v.w) << 16);
      *(uint2*)&Alds[row * 128 + c4 * 4] = pk;
    }
  }
  __syncthreads();

  const int wave = tid >> 5;       // 0..7 -> M rows [wave*16, wave*16+16)
  const int lane = tid & 31;
  const int half = lane >> 4;      // 0 or 1
  const int l16  = lane & 15;
  const int aRow = wave * 16 + l16;

  // 8 accumulator tiles (wave's 16x128 strip)
  v8f acc[8];
#pragma unroll
  for (int t = 0; t < 8; ++t) acc[t] = (v8f){0.f,0.f,0.f,0.f,0.f,0.f,0.f,0.f};

  // K-chunk outer; A fragment loaded once per chunk, reused by 8 WMMAs
#pragma unroll
  for (int c = 0; c < 4; ++c) {
    ABfrag af;
    const int ak = c * 32;
#pragma unroll
    for (int v = 0; v < 8; ++v) {
      int k0 = ak + 2 * v + half * 16;
      af.u[v] = *(const unsigned int*)&Alds[aRow * 128 + k0];
    }
    const int bk = c * 32 + half * 16;
#pragma unroll
    for (int t = 0; t < 8; ++t) {
      ABfrag bfr;
      const int bn = t * 16 + l16;
#pragma unroll
      for (int v = 0; v < 8; ++v)
        bfr.u[v] = *(const unsigned int*)&Blds[bn * 128 + bk + 2 * v];
      acc[t] = __builtin_amdgcn_wmma_f32_16x16x32_bf16(
          false, af.v, false, bfr.v, (short)0, acc[t], false, false);
    }
  }

  // Row-major epilogue: one address + one guard per output row
  const int rowBase = m0 + wave * 16 + half * 8;
#pragma unroll
  for (int r = 0; r < 8; ++r) {
    const int gr = rowBase + r;
    if (gr < M) {
      float* cp = C + (size_t)gr * 128 + l16;
      if (MODE == 1) {
        const float* ap = addTab + (size_t)addIdx[gr] * 128 + l16;
#pragma unroll
        for (int t = 0; t < 8; ++t) cp[t * 16] = acc[t][r] + ap[t * 16];
      } else {
#pragma unroll
        for (int t = 0; t < 8; ++t) cp[t * 16] = acc[t][r];
      }
    }
  }
}

// ---------------------------------------------------------------------------
// Weight prep: 6 matrices of [128,128] fp32 -> bf16, transposed to [n][k]
//   m=0: W10   m=1: W2_10[:128]  m=2: W2_10[128:]
//   m=3: W11   m=4: W2_11[:128]  m=5: W2_11[128:]
// ---------------------------------------------------------------------------
__global__ void prep_weights(const float* __restrict__ W10,
                             const float* __restrict__ W2_10,
                             const float* __restrict__ W11,
                             const float* __restrict__ W2_11,
                             unsigned short* __restrict__ BtAll)
{
  int i = blockIdx.x * blockDim.x + threadIdx.x;
  if (i >= 6 * 16384) return;
  int m = i / 16384, j = i % 16384;
  int n = j >> 7, k = j & 127;
  float v;
  switch (m) {
    case 0: v = W10[k * 128 + n]; break;
    case 1: v = W2_10[k * 128 + n]; break;
    case 2: v = W2_10[(128 + k) * 128 + n]; break;
    case 3: v = W11[k * 128 + n]; break;
    case 4: v = W2_11[k * 128 + n]; break;
    default: v = W2_11[(128 + k) * 128 + n]; break;
  }
  BtAll[(size_t)m * 16384 + (size_t)n * 128 + k] = f32_to_bf16(v);
}

// ---------------------------------------------------------------------------
// Small elementwise / scatter kernels
// ---------------------------------------------------------------------------
__global__ void fill_f32(float* p, float v, long n) {
  long i = (long)blockIdx.x * blockDim.x + threadIdx.x;
  if (i < n) p[i] = v;
}

__global__ void deg_count(const int* __restrict__ src, float* deg, int E) {
  int e = blockIdx.x * blockDim.x + threadIdx.x;
  if (e < E) atomicAdd(&deg[src[e]], 1.f);
}

__global__ void dinv_inplace(float* deg, int N) {
  int i = blockIdx.x * blockDim.x + threadIdx.x;
  if (i < N) deg[i] = rsqrtf(deg[i] + 1.f);
}

__global__ void row_scale(const float* __restrict__ h, const float* __restrict__ w,
                          float* __restrict__ out, long n) {
  long i = (long)blockIdx.x * blockDim.x + threadIdx.x;
  if (i < n) out[i] = h[i] * w[i & 127];
}

__global__ void gcn_scatter(const float* __restrict__ hs,
                            const int* __restrict__ src, const int* __restrict__ dst,
                            const float* __restrict__ dinv, float* out, int E) {
  long i = (long)blockIdx.x * blockDim.x + threadIdx.x;
  if (i >= (long)E * 128) return;
  int e = (int)(i >> 7), k = (int)(i & 127);
  int s = src[e], d = dst[e];
  float nrm = dinv[s] * dinv[d];
  atomicAdd(&out[(size_t)d * 128 + k], nrm * hs[(size_t)s * 128 + k]);
}

__global__ void gcn_finalize(float* out, const float* __restrict__ hs,
                             const float* __restrict__ dinv, const float* __restrict__ b,
                             int N, int doRelu) {
  long i = (long)blockIdx.x * blockDim.x + threadIdx.x;
  if (i >= (long)N * 128) return;
  int n = (int)(i >> 7), k = (int)(i & 127);
  float di = dinv[n];
  float v = out[i] + di * di * hs[i] + b[k];
  out[i] = doRelu ? fmaxf(v, 0.f) : v;
}

__global__ void concat_rows(const float* __restrict__ x, const float* __restrict__ r,
                            float* __restrict__ xr, long nx, long ntot128) {
  long i = (long)blockIdx.x * blockDim.x + threadIdx.x;
  if (i >= ntot128) return;
  xr[i] = (i < nx) ? x[i] : r[i - nx];
}

__global__ void relu_copy(const float* __restrict__ in, float* __restrict__ out, long n) {
  long i = (long)blockIdx.x * blockDim.x + threadIdx.x;
  if (i < n) out[i] = fmaxf(in[i], 0.f);
}

// ---------------------------------------------------------------------------
// Attention: one wave (32 lanes) per edge for the 128-d dot products
// ---------------------------------------------------------------------------
__global__ __launch_bounds__(256) void attn_alpha(
    const float* __restrict__ hW, const float* __restrict__ xj,
    const int* __restrict__ dst, const float* __restrict__ att,
    float* __restrict__ alpha, float* amax, int E)
{
  int e = (blockIdx.x * 256 + threadIdx.x) >> 5;
  int lane = threadIdx.x & 31;
  if (e >= E) return;
  int d = dst[e];
  const float4* hp = (const float4*)(hW + (size_t)d * 128);
  const float4* xp = (const float4*)(xj + (size_t)e * 128);
  const float4* al = (const float4*)att;
  float4 h4 = hp[lane], x4 = xp[lane];
  float4 a4 = al[lane], b4 = al[32 + lane];
  float s = h4.x * a4.x + h4.y * a4.y + h4.z * a4.z + h4.w * a4.w
          + x4.x * b4.x + x4.y * b4.y + x4.z * b4.z + x4.w * b4.w;
  for (int off = 16; off; off >>= 1) s += __shfl_xor(s, off, 32);
  if (lane == 0) {
    float a = (s > 0.f) ? s : 0.2f * s;   // leaky_relu(0.2)
    alpha[e] = a;
    atomicMaxFloat(&amax[d], a);
  }
}

__global__ void attn_exp(float* alpha_ev, const float* __restrict__ amax,
                         const int* __restrict__ dst, float* denom, int E) {
  int e = blockIdx.x * blockDim.x + threadIdx.x;
  if (e >= E) return;
  int d = dst[e];
  float ev = expf(alpha_ev[e] - amax[d]);
  alpha_ev[e] = ev;
  atomicAdd(&denom[d], ev);
}

__global__ void attn_msg(const float* __restrict__ xj, const float* __restrict__ r_ij,
                         const float* __restrict__ sgn, const float* __restrict__ ev,
                         const float* __restrict__ denom, const int* __restrict__ dst,
                         float* out, int E) {
  long i = (long)blockIdx.x * blockDim.x + threadIdx.x;
  if (i >= (long)E * 128) return;
  int e = (int)(i >> 7);
  int d = dst[e];
  float w = ev[e] / denom[d];
  float val = (xj[i] + r_ij[i] * sgn[e]) * w;
  atomicAdd(&out[(size_t)d * 128 + (i & 127)], val);
}

// ---------------------------------------------------------------------------
// L2 normalize rows of [N,128]; one wave per row
// ---------------------------------------------------------------------------
__global__ __launch_bounds__(256) void l2norm_rows(const float* __restrict__ in,
                                                   float* __restrict__ out, int N) {
  int row = (blockIdx.x * 256 + threadIdx.x) >> 5;
  int lane = threadIdx.x & 31;
  if (row >= N) return;
  const float4* p = (const float4*)(in + (size_t)row * 128);
  float4 v = p[lane];
  float s = v.x * v.x + v.y * v.y + v.z * v.z + v.w * v.w;
  for (int off = 16; off; off >>= 1) s += __shfl_xor(s, off, 32);
  float inv = 1.0f / fmaxf(sqrtf(s), 1e-12f);
  float4 o = make_float4(v.x * inv, v.y * inv, v.z * inv, v.w * inv);
  ((float4*)(out + (size_t)row * 128))[lane] = o;
}

// ---------------------------------------------------------------------------
// Host orchestration
// ---------------------------------------------------------------------------
static inline float* carve(char*& p, size_t nfloats) {
  float* r = (float*)p;
  p += ((nfloats * sizeof(float) + 255) / 256) * 256;
  return r;
}
static inline int blk(long n, int t) { return (int)((n + t - 1) / t); }

extern "C" void kernel_launch(void* const* d_in, const int* in_sizes, int n_in,
                              void* d_out, int out_size, void* d_ws, size_t ws_size,
                              hipStream_t stream) {
  const float* x    = (const float*)d_in[0];
  const int*   ei   = (const int*)d_in[1];
  const float* rrel = (const float*)d_in[2];
  const float* sgn  = (const float*)d_in[3];
  const float* rij  = (const float*)d_in[4];
  const float* w2   = (const float*)d_in[5];
  const float* b2   = (const float*)d_in[6];
  const float* w3   = (const float*)d_in[7];
  const float* b3   = (const float*)d_in[8];
  const float* W10  = (const float*)d_in[9];
  const float* W2_10= (const float*)d_in[10];
  const float* att10= (const float*)d_in[11];
  const float* W11  = (const float*)d_in[12];
  const float* W2_11= (const float*)d_in[13];
  const float* att11= (const float*)d_in[14];
  float* out = (float*)d_out;

  const int N = in_sizes[0] / 128;
  const int R = in_sizes[2] / 128;
  const int E = in_sizes[3];
  const int Ntot = N + R;
  const int* src = ei;
  const int* dst = ei + E;

  // ---- workspace carve ----
  char* p = (char*)d_ws;
  float* deg    = carve(p, (size_t)N);              // deg -> dinv in place
  float* hs     = carve(p, (size_t)N * 128);
  float* g1     = carve(p, (size_t)N * 128);
  float* g2     = carve(p, (size_t)N * 128);
  float* xr     = carve(p, (size_t)Ntot * 128);
  float* xr1    = carve(p, (size_t)Ntot * 128);
  float* hW     = carve(p, (size_t)Ntot * 128);
  float* hW2a   = carve(p, (size_t)Ntot * 128);
  float* outacc = carve(p, (size_t)Ntot * 128);
  float* amax   = carve(p, (size_t)Ntot);
  float* denom  = carve(p, (size_t)Ntot);
  float* aev    = carve(p, (size_t)E);              // alpha -> ev in place
  float* xj     = carve(p, (size_t)E * 128);
  unsigned short* BtAll = (unsigned short*)carve(p, (size_t)(6 * 16384 + 1) / 2);
  (void)ws_size; (void)n_in; (void)out_size;

  const long NE128 = (long)E * 128;
  const long NN128 = (long)N * 128;
  const long NT128 = (long)Ntot * 128;

  // ---- weight prep (bf16, transposed) ----
  prep_weights<<<blk(6 * 16384, 256), 256, 0, stream>>>(W10, W2_10, W11, W2_11, BtAll);

  // ================= branch 1: two plain GCN layers =================
  hipMemsetAsync(deg, 0, (size_t)N * 4, stream);
  deg_count<<<blk(E, 256), 256, 0, stream>>>(src, deg, E);
  dinv_inplace<<<blk(N, 256), 256, 0, stream>>>(deg, N);

  row_scale<<<blk(NN128, 256), 256, 0, stream>>>(x, w2, hs, NN128);
  hipMemsetAsync(g1, 0, (size_t)NN128 * 4, stream);
  gcn_scatter<<<blk(NE128, 256), 256, 0, stream>>>(hs, src, dst, deg, g1, E);
  gcn_finalize<<<blk(NN128, 256), 256, 0, stream>>>(g1, hs, deg, b2, N, 1);

  row_scale<<<blk(NN128, 256), 256, 0, stream>>>(g1, w3, hs, NN128);
  hipMemsetAsync(g2, 0, (size_t)NN128 * 4, stream);
  gcn_scatter<<<blk(NE128, 256), 256, 0, stream>>>(hs, src, dst, deg, g2, E);
  gcn_finalize<<<blk(NN128, 256), 256, 0, stream>>>(g2, hs, deg, b3, N, 0);

  l2norm_rows<<<blk((long)N * 32, 256), 256, 0, stream>>>(g2, out, N);

  // ================= branch 2: relation-attention layers =================
  concat_rows<<<blk(NT128, 256), 256, 0, stream>>>(x, rrel, xr, NN128, NT128);

  const int gN = blk(Ntot, 128), gE = blk(E, 128);
  for (int layer = 0; layer < 2; ++layer) {
    const float* h   = (layer == 0) ? xr : xr1;
    const float* att = (layer == 0) ? att10 : att11;
    const unsigned short* BtW   = BtAll + (size_t)(layer ? 3 : 0) * 16384;
    const unsigned short* Bt2lo = BtAll + (size_t)(layer ? 4 : 1) * 16384;
    const unsigned short* Bt2hi = BtAll + (size_t)(layer ? 5 : 2) * 16384;

    // hW = h @ W ; hW2a = h @ W2[:d]
    gemm128_bf16<0><<<gN, 256, 0, stream>>>(h, BtW,   hW,   Ntot, nullptr, nullptr, nullptr);
    gemm128_bf16<0><<<gN, 256, 0, stream>>>(h, Bt2lo, hW2a, Ntot, nullptr, nullptr, nullptr);
    // xj = (r_ij * s) @ W2[d:] + hW2a[src]   (A-scale + gather-epilogue fused)
    gemm128_bf16<1><<<gE, 256, 0, stream>>>(rij, Bt2hi, xj, E, sgn, hW2a, src);

    fill_f32<<<blk(Ntot, 256), 256, 0, stream>>>(amax, -INFINITY, Ntot);
    hipMemsetAsync(denom, 0, (size_t)Ntot * 4, stream);
    hipMemsetAsync(outacc, 0, (size_t)NT128 * 4, stream);

    attn_alpha<<<blk((long)E * 32, 256), 256, 0, stream>>>(hW, xj, dst, att, aev, amax, E);
    attn_exp<<<blk(E, 256), 256, 0, stream>>>(aev, amax, dst, denom, E);
    attn_msg<<<blk(NE128, 256), 256, 0, stream>>>(xj, rij, sgn, aev, denom, dst, outacc, E);

    if (layer == 0)
      relu_copy<<<blk(NT128, 256), 256, 0, stream>>>(outacc, xr1, NT128);
    else
      l2norm_rows<<<blk((long)N * 32, 256), 256, 0, stream>>>(outacc, out + NN128, N);
  }
}
